// InductiveBiasLoss_18545668784407
// MI455X (gfx1250) — matrix operand order repaired and checked by reading, not compiled, not run on hardware
//
#include <hip/hip_runtime.h>
#include <math.h>

typedef __attribute__((ext_vector_type(16))) _Float16 v16h;
typedef __attribute__((ext_vector_type(8)))  float    v8f;

#define F_LEN  20.0f
#define PIXSZ  0.03f
#define IMG    256
#define NPIX   (IMG*IMG)
#define NBATCH 64
#define VALID_DIM 246   // 256 - 11 + 1

// ---------------------------------------------------------------- pose -> R,t
__global__ void pose_kernel(const float* __restrict__ pose, float* __restrict__ rt) {
  int b = threadIdx.x;
  if (b >= NBATCH) return;
  const float* p = pose + b * 6;
  float t0 = p[0], t1 = p[1], t2 = p[2];
  float cx = cosf(p[3]), sx = sinf(p[3]);
  float cy = cosf(p[4]), sy = sinf(p[4]);
  float cz = cosf(p[5]), sz = sinf(p[5]);
  float* o = rt + b * 12;           // R = Rz @ Ry @ Rx
  o[0] = cz*cy;  o[1] = cz*sy*sx - sz*cx;  o[2] = cz*sy*cx + sz*sx;
  o[3] = sz*cy;  o[4] = sz*sy*sx + cz*cx;  o[5] = sz*sy*cx - cz*sx;
  o[6] = -sy;    o[7] = cy*sx;             o[8] = cy*cx;
  o[9] = t0; o[10] = t1; o[11] = t2;
}

// --------------------------- fused depth->point->transform->project->sample
__global__ void warp_kernel(const float* __restrict__ im_pred,
                            const float* __restrict__ im_xf,
                            const float* __restrict__ depth,
                            const float* __restrict__ rt,
                            float* __restrict__ geo_out,
                            float* __restrict__ imx_out) {
  int b   = blockIdx.y;
  int pix = blockIdx.x * blockDim.x + threadIdx.x;
  if (pix >= NPIX) return;
  int py = pix >> 8, px = pix & 255;
  const float* R = rt + b * 12;

  float Z  = depth[b * NPIX + pix];
  float xs = ((float)px - 127.5f) * PIXSZ;
  float ys = ((float)py - 127.5f) * PIXSZ;
  float X  = xs * Z / F_LEN;
  float Y  = ys * Z / F_LEN;
  float Xp = R[0]*X + R[1]*Y + R[2]*Z + R[9];
  float Yp = R[3]*X + R[4]*Y + R[5]*Z + R[10];
  float Zp = R[6]*X + R[7]*Y + R[8]*Z + R[11];
  float Zs = (fabsf(Zp) > 1e-6f) ? Zp : 1e-6f;
  float xn = (F_LEN * Xp / Zs) / (PIXSZ * 127.5f);
  float yn = (F_LEN * Yp / Zs) / (PIXSZ * 127.5f);

  float gx = (xn + 1.0f) * 0.5f * (float)IMG - 0.5f;
  float gy = (yn + 1.0f) * 0.5f * (float)IMG - 0.5f;
  float x0 = floorf(gx), y0 = floorf(gy);
  float wx1 = gx - x0, wy1 = gy - y0;
  float wx0 = 1.0f - wx1, wy0 = 1.0f - wy1;

  const float* img = im_pred + b * NPIX;
  float fx[2] = {x0, x0 + 1.0f}, fy[2] = {y0, y0 + 1.0f};
  float wx[2] = {wx0, wx1},      wy[2] = {wy0, wy1};
  float acc = 0.0f;
  #pragma unroll
  for (int j = 0; j < 2; ++j)
    #pragma unroll
    for (int i = 0; i < 2; ++i) {
      float ix = fx[i], iy = fy[j];
      bool valid = (ix >= 0.0f) && (ix < (float)IMG) && (iy >= 0.0f) && (iy < (float)IMG);
      int ixc = (int)fminf(fmaxf(ix, 0.0f), 255.0f);
      int iyc = (int)fminf(fmaxf(iy, 0.0f), 255.0f);
      float v = img[iyc * IMG + ixc];
      acc += valid ? v * (wx[i] * wy[j]) : 0.0f;
    }

  geo_out[b * NPIX + pix] = acc;
  imx_out[b * NPIX + pix] = (acc > 0.0f) ? im_xf[b * NPIX + pix] : 0.0f;
}

// A-matrix f16 16x32: K index held by vector element e for lane-half `half`
__device__ __forceinline__ int a_k(int e, int half) {
  int vg = e >> 1, sub = e & 1;
  return ((vg < 4) ? 0 : 16) + half * 8 + (vg & 3) * 2 + sub;
}

// -------------------- fused SSIM: separable 11-tap Gaussian as WMMA matmuls
__global__ void __launch_bounds__(32)
ssim_kernel(const float* __restrict__ geo,   // b-image (im_pred_geo)
            const float* __restrict__ imx,   // a-image (im_x)
            float* __restrict__ partials) {
  __shared__ _Float16 P[5][32][32];  // product maps (26x26 valid, zero pad)
  __shared__ _Float16 Hs[5][32][16]; // horizontally filtered (26 valid rows)

  int lane = threadIdx.x;
  int half = lane >> 4;
  int lid  = lane & 15;
  int x0 = blockIdx.x * 16, y0 = blockIdx.y * 16, b = blockIdx.z;

  // normalized 11-tap gaussian, sigma = 1.5
  float g[11];
  {
    float s = 0.0f;
    #pragma unroll
    for (int k = 0; k < 11; ++k) { float d = (float)(k - 5); g[k] = expf(-d * d / 4.5f); s += g[k]; }
    float inv = 1.0f / s;
    #pragma unroll
    for (int k = 0; k < 11; ++k) g[k] *= inv;
  }

  // zero the padded product planes
  for (int i = lane; i < 5 * 32 * 32; i += 32) ((_Float16*)P)[i] = (_Float16)0.0f;
  __syncthreads();

  // load 26x26 halo tile, build 5 product maps (data stays in [0,1] -> f16 safe)
  const float* aim = imx + b * NPIX;
  const float* bim = geo + b * NPIX;
  for (int idx = lane; idx < 26 * 26; idx += 32) {
    int r = idx / 26, c = idx - r * 26;
    int gy = min(y0 + r, 255), gx = min(x0 + c, 255);
    float av = aim[gy * IMG + gx];
    float bv = bim[gy * IMG + gx];
    P[0][r][c] = (_Float16)av;
    P[1][r][c] = (_Float16)bv;
    P[2][r][c] = (_Float16)(av * av);
    P[3][r][c] = (_Float16)(bv * bv);
    P[4][r][c] = (_Float16)(av * bv);
  }
  __syncthreads();

  // banded Gaussian weight matrices (constants, built per lane)
  v16h Bh;  // horizontal B [K=src col r][N=out col x] = g[r-x]
  v16h Av;  // vertical   A [M=out row y][K=src row r] = g[r-y]
  #pragma unroll
  for (int e = 0; e < 16; ++e) {
    int kb = a_k(e, half);            // B layout: interleaved K pattern
    int db = kb - lid;
    Bh[e] = (_Float16)((db >= 0 && db < 11) ? g[db] : 0.0f);
    int ka = a_k(e, half);            // A layout: interleaved K pattern
    int da = ka - lid;
    Av[e] = (_Float16)((da >= 0 && da < 11) ? g[da] : 0.0f);
  }

  // horizontal pass: H[26x16] = P[26x32] x Gcol[32x16]  (rows split 16+16)
  #pragma unroll
  for (int c = 0; c < 5; ++c) {
    #pragma unroll
    for (int h = 0; h < 2; ++h) {
      v16h A;
      #pragma unroll
      for (int e = 0; e < 16; ++e)
        A[e] = P[c][h * 16 + lid][a_k(e, half)];
      v8f acc = {};
      acc = __builtin_amdgcn_wmma_f32_16x16x32_f16(false, A, false, Bh,
                                                   (short)0, acc, false, false);
      #pragma unroll
      for (int i = 0; i < 8; ++i)
        Hs[c][h * 16 + i + 8 * half][lid] = (_Float16)acc[i];
    }
  }
  __syncthreads();

  // vertical pass: Out[16x16] = Grow[16x32] x H[32x16]
  v8f O[5];
  #pragma unroll
  for (int c = 0; c < 5; ++c) {
    v16h Bv;
    #pragma unroll
    for (int e = 0; e < 16; ++e)
      Bv[e] = Hs[c][a_k(e, half)][lid];
    v8f acc = {};
    acc = __builtin_amdgcn_wmma_f32_16x16x32_f16(false, Av, false, Bv,
                                                 (short)0, acc, false, false);
    O[c] = acc;
  }

  // SSIM map (scale-invariant form: data in [0,1], C scaled by 255^-2)
  const float C1 = 1e-4f, C2 = 9e-4f;
  float lsum = 0.0f;
  #pragma unroll
  for (int i = 0; i < 8; ++i) {
    int oy = y0 + i + 8 * half;
    int ox = x0 + lid;
    float mu1 = O[0][i], mu2 = O[1][i];
    float s1  = fmaxf(O[2][i] - mu1 * mu1, 0.0f);
    float s2  = fmaxf(O[3][i] - mu2 * mu2, 0.0f);
    float s12 = O[4][i] - mu1 * mu2;
    float m = ((2.0f * mu1 * mu2 + C1) * (2.0f * s12 + C2)) /
              ((mu1 * mu1 + mu2 * mu2 + C1) * (s1 + s2 + C2));
    lsum += (oy < VALID_DIM && ox < VALID_DIM) ? m : 0.0f;
  }

  #pragma unroll
  for (int off = 16; off > 0; off >>= 1)
    lsum += __shfl_xor(lsum, off, 32);
  if (lane == 0)
    partials[blockIdx.x + 16 * blockIdx.y + 256 * blockIdx.z] = lsum;
}

// ------------------------------------------- deterministic final reduction
__global__ void finish_kernel(const float* __restrict__ partials, float* __restrict__ out0) {
  __shared__ float red[256];
  int t = threadIdx.x;
  float s = 0.0f;
  for (int i = t; i < 16 * 16 * NBATCH; i += 256) s += partials[i];
  red[t] = s;
  __syncthreads();
  for (int w = 128; w > 0; w >>= 1) {
    if (t < w) red[t] += red[t + w];
    __syncthreads();
  }
  if (t == 0) {
    float mean = red[0] / (float)(NBATCH * VALID_DIM * VALID_DIM);
    out0[0] = expf(-mean);
  }
}

extern "C" void kernel_launch(void* const* d_in, const int* in_sizes, int n_in,
                              void* d_out, int out_size, void* d_ws, size_t ws_size,
                              hipStream_t stream) {
  const float* pose    = (const float*)d_in[0];
  const float* im_pred = (const float*)d_in[1];
  const float* im_xf   = (const float*)d_in[2];
  const float* depth   = (const float*)d_in[3];
  // d_in[4] (intrinsics) is unused by the reference.

  float* out = (float*)d_out;
  float* geo = out + 1;                  // im_pred_geo
  float* imx = out + 1 + NBATCH * NPIX;  // im_x

  float* rt       = (float*)d_ws;          // 64 * 12 floats
  float* partials = (float*)d_ws + 1024;   // 16384 floats

  pose_kernel<<<1, 64, 0, stream>>>(pose, rt);
  warp_kernel<<<dim3(NPIX / 256, NBATCH), 256, 0, stream>>>(im_pred, im_xf, depth, rt, geo, imx);
  ssim_kernel<<<dim3(16, 16, NBATCH), 32, 0, stream>>>(geo, imx, partials);
  finish_kernel<<<1, 256, 0, stream>>>(partials, out);
}
